// DyGraphConv2d_11922829214268
// MI455X (gfx1250) — matrix operand order repaired and checked by reading, not compiled, not run on hardware
//
#include <hip/hip_runtime.h>
#include <math.h>

#define BATCH 32
#define CCH   192
#define NND   784      // H*W = 28*28
#define OUTC  384
#define KNN   9
#define CTILES 49      // 784/16

typedef float    v2f  __attribute__((ext_vector_type(2)));
typedef float    v8f  __attribute__((ext_vector_type(8)));
typedef _Float16 v8h  __attribute__((ext_vector_type(8)));
typedef _Float16 v16h __attribute__((ext_vector_type(16)));

__device__ __forceinline__ float gelu_exact(float v) {
    return 0.5f * v * (1.0f + erff(v * 0.70710678118654752440f));
}

// ---------------------------------------------------------------------------
// Kernel A: per-node L2 normalization -> f16, node-major layout [b][n][c].
// Normalized values are in [-1,1]: ideal f16 conditioning for the Gram GEMM.
// ---------------------------------------------------------------------------
__global__ __launch_bounds__(256) void normalize_kernel(const float* __restrict__ x,
                                                        _Float16* __restrict__ xnH) {
    int b = blockIdx.x >> 2;                 // 4 blocks of 256 per batch
    int n = (blockIdx.x & 3) * 256 + threadIdx.x;
    if (n >= NND) return;
    const float* xb = x + (size_t)b * CCH * NND;
    float s = 0.f;
    for (int c = 0; c < CCH; ++c) { float v = xb[c * NND + n]; s += v * v; }
    float inv = 1.0f / fmaxf(sqrtf(s), 1e-12f);
    _Float16* ob = xnH + ((size_t)b * NND + n) * CCH;
    for (int c = 0; c < CCH; ++c) ob[c] = (_Float16)(xb[c * NND + n] * inv);
}

// ---------------------------------------------------------------------------
// Kernel B: Gram row-block (16 x 784) via f16 WMMA 16x16x32 (f32 accum),
// then top-9 per row. One workgroup (8 waves) per (batch, 16-row block).
// All 6 B fragments are loaded into registers first (one 12-wide load
// clause, single loadcnt wait), then 6 WMMAs issue back-to-back.
// ---------------------------------------------------------------------------
__global__ __launch_bounds__(256) void gram_topk_kernel(const _Float16* __restrict__ xnH,
                                                        int* __restrict__ nn) {
    __shared__ float Gs[16][NND];   // 16 x 784 Gram rows (49 KB)

    int b  = blockIdx.x / CTILES;
    int r0 = (blockIdx.x % CTILES) * 16;
    const _Float16* xb = xnH + (size_t)b * NND * CCH;

    int w = threadIdx.x >> 5, lane = threadIdx.x & 31;
    int ml = lane & 15, hi = lane >> 4;

    // A 16x32 f16 frag layout: lane<16 holds K={kk..kk+7, kk+16..kk+23},
    // lane>=16 holds K={kk+8..kk+15, kk+24..kk+31}  -> base shift 8*hi.
    union U { v16h v; v8h h[2]; };
    U afrag[6];
    const _Float16* arow = xb + (size_t)(r0 + ml) * CCH;
#pragma unroll
    for (int c = 0; c < 6; ++c) {
        int kk = c * 32;
        afrag[c].h[0] = *(const v8h*)(arow + kk + 8 * hi);
        afrag[c].h[1] = *(const v8h*)(arow + kk + 16 + 8 * hi);
    }

    for (int ct = w; ct < CTILES; ct += 8) {
        int c0 = ct * 16;
        // Branch-free prefetch of the next tile's B rows (clamped index).
        int pf = ct + 8 < CTILES ? (ct + 8) * 16 : (CTILES - 1) * 16;
        __builtin_prefetch(xb + (size_t)(pf + ml) * CCH + 32 * hi, 0, 1);

        // B 32x16 f16 frag: lanes 0-15 hold K={kk..kk+7, kk+16..kk+23} of column n=ml,
        // lanes 16-31 hold K={kk+8..kk+15, kk+24..kk+31} -> base shift 8*hi.
        const _Float16* brow = xb + (size_t)(c0 + ml) * CCH + 8 * hi;
        U bu[6];
#pragma unroll
        for (int c = 0; c < 6; ++c) {          // 12-load clause, one wait
            bu[c].h[0] = *(const v8h*)(brow + c * 32);
            bu[c].h[1] = *(const v8h*)(brow + c * 32 + 16);
        }
        v8f acc = {};
#pragma unroll
        for (int c = 0; c < 6; ++c)            // back-to-back accumulation
            acc = __builtin_amdgcn_wmma_f32_16x16x32_f16(false, afrag[c].v, false, bu[c].v,
                                                         (short)0, acc, false, false);
        // C/D layout: element (M,N): lane = N + 16*(M>=8), vgpr = M&7
#pragma unroll
        for (int v = 0; v < 8; ++v) Gs[v + 8 * hi][c0 + ml] = acc[v];
    }
    __syncthreads();

    // Top-9 per row. Wave w handles rows w and w+8. 9 cooperative argmax passes.
    for (int rr = 0; rr < 2; ++rr) {
        int row = w + rr * 8;
        for (int pass = 0; pass < KNN; ++pass) {
            float best = -INFINITY; int bidx = 0;
            for (int c = lane; c < NND; c += 32) {
                float v = Gs[row][c];
                if (v > best) { best = v; bidx = c; }
            }
#pragma unroll
            for (int off = 16; off > 0; off >>= 1) {
                float ov = __shfl_xor(best, off, 32);
                int   oi = __shfl_xor(bidx, off, 32);
                if (ov > best || (ov == best && oi < bidx)) { best = ov; bidx = oi; }
            }
            if (lane == 0) {
                nn[((size_t)b * NND + r0 + row) * KNN + pass] = bidx;
                Gs[row][bidx] = -INFINITY;     // remove winner
            }
            __syncthreads();                   // uniform across all 8 waves
        }
    }
}

// ---------------------------------------------------------------------------
// Kernel C: per-node projection u[b][n][o] = W_g . x_part (4 group GEMMs),
// kept in f32 WMMA 16x16x4 since it feeds the f32 output directly.
// One wave per 16x16 output tile, K=96 -> 24 WMMA steps.
// ---------------------------------------------------------------------------
__global__ __launch_bounds__(256) void proj_kernel(const float* __restrict__ x,
                                                   const float* __restrict__ wgt,
                                                   float* __restrict__ u) {
    int w = threadIdx.x >> 5, lane = threadIdx.x & 31;
    int ml = lane & 15, hi = lane >> 4;
    int t  = blockIdx.x * 8 + w;               // 37632 tiles total
    int b  = t / 1176;  int r  = t % 1176;     // 4 groups * 6 row tiles * 49 col tiles
    int g  = r / 294;   int r2 = r % 294;
    int ot = r2 / CTILES; int ct = r2 % CTILES;
    int c0 = ct * 16;

    // groups 0,2 use x channels 0..95 ; groups 1,3 use channels 96..191
    const float* xb = x + (size_t)b * CCH * NND + (size_t)(g & 1) * 96 * NND;
    const float* wr = wgt + (size_t)(g * 96 + ot * 16) * 96;

    v8f acc = {};
#pragma unroll 4
    for (int kk = 0; kk < 96; kk += 4) {
        v2f a = *(const v2f*)&wr[ml * 96 + kk + 2 * hi];   // W[o=ml][kk..]
        v2f bv;
        bv.x = xb[(kk + 2 * hi    ) * NND + c0 + ml];
        bv.y = xb[(kk + 2 * hi + 1) * NND + c0 + ml];
        acc = __builtin_amdgcn_wmma_f32_16x16x4_f32(false, a, false, bv,
                                                    (short)0, acc, false, false);
    }
    int n = c0 + ml;
    float* ur = u + ((size_t)b * NND + n) * OUTC + g * 96 + ot * 16 + 8 * hi;
#pragma unroll
    for (int v = 0; v < 8; ++v) ur[v] = acc[v];            // 8 contiguous floats/lane
}

// ---------------------------------------------------------------------------
// Kernel D: fused bias + exact GELU + max over K neighbors.
// ch < 192: k-independent (x_i part) -> gelu(u + bias).
// ch >=192: max_k gelu(u[j_k] - u[i] + bias)  (x_j - x_i part).
// ---------------------------------------------------------------------------
__global__ __launch_bounds__(384) void out_kernel(const float* __restrict__ u,
                                                  const int* __restrict__ nn,
                                                  const float* __restrict__ bias,
                                                  float* __restrict__ out) {
    __shared__ int js[KNN];
    int b = blockIdx.x / NND;
    int n = blockIdx.x % NND;
    int o = threadIdx.x;
    if (o < KNN) js[o] = nn[((size_t)b * NND + n) * KNN + o];
    __syncthreads();

    const float* ub = u + (size_t)b * NND * OUTC;
    float self = ub[(size_t)n * OUTC + o];
    float bi   = bias[o];
    float r;
    if (o < 192) {
        r = gelu_exact(self + bi);
    } else {
        float m = -INFINITY;
#pragma unroll
        for (int k = 0; k < KNN; ++k) {
            float vj = ub[(size_t)js[k] * OUTC + o];       // coalesced over o
            m = fmaxf(m, gelu_exact(vj - self + bi));
        }
        r = m;
    }
    out[((size_t)b * OUTC + o) * NND + n] = r;
}

// ---------------------------------------------------------------------------
extern "C" void kernel_launch(void* const* d_in, const int* in_sizes, int n_in,
                              void* d_out, int out_size, void* d_ws, size_t ws_size,
                              hipStream_t stream) {
    const float* x    = (const float*)d_in[0];   // [32,192,28,28]
    const float* wgt  = (const float*)d_in[1];   // [384,96]
    const float* bias = (const float*)d_in[2];   // [384]
    float* out = (float*)d_out;                  // [32,384,28,28]

    // Workspace: xnH (B*N*C f16, 9.6MB) | nn_idx (B*N*9 i32, 0.9MB) | u (B*N*384 f32, 38.5MB)
    _Float16* xnH = (_Float16*)d_ws;
    int*      nn  = (int*)(xnH + (size_t)BATCH * NND * CCH);
    float*    u   = (float*)(nn + (size_t)BATCH * NND * KNN);

    hipLaunchKernelGGL(normalize_kernel, dim3(BATCH * 4),      dim3(256), 0, stream, x, xnH);
    hipLaunchKernelGGL(gram_topk_kernel, dim3(BATCH * CTILES), dim3(256), 0, stream, xnH, nn);
    hipLaunchKernelGGL(proj_kernel,      dim3(4704),           dim3(256), 0, stream, x, wgt, u);
    hipLaunchKernelGGL(out_kernel,       dim3(BATCH * NND),    dim3(384), 0, stream, u, nn, bias, out);
}
